// AttnEmbed_82918638616989
// MI455X (gfx1250) — matrix operand order repaired and checked by reading, not compiled, not run on hardware
//
#include <hip/hip_runtime.h>

typedef __attribute__((ext_vector_type(16))) __bf16 v16bf;
typedef __attribute__((ext_vector_type(2)))  __bf16 bf16x2;
typedef __attribute__((ext_vector_type(8)))  float  v8f;
typedef __attribute__((ext_vector_type(2)))  float  f32x2;
typedef __attribute__((ext_vector_type(4)))  float  f32x4;

#define BB 32
#define TT 1024
#define JJ 128
#define DD 256
#define TPITCH 130   // LDS pitch (elements): even (4B-aligned pairs), 65 dwords -> conflict-free

// ---------------------------------------------------------------------------
// k0: s_ctx[b,t] = ctx[b,t,:]·w1 ; s_qry[b,j] = query[b,j,:]·w2
// one wave per row, lane-parallel over D
// ---------------------------------------------------------------------------
__global__ __launch_bounds__(256) void k0_rowdots(
    const float* __restrict__ ctx, const float* __restrict__ qry,
    const float* __restrict__ w, float* __restrict__ sctx, float* __restrict__ sqry)
{
    const int gw = (blockIdx.x * 256 + threadIdx.x) >> 5;
    const int l  = threadIdx.x & 31;
    const int NC = BB * TT;
    const int NQ = BB * JJ;
    if (gw < NC) {
        const float* row = ctx + (size_t)gw * DD;
        float s = 0.f;
#pragma unroll
        for (int k = 0; k < 8; ++k) s += row[l + 32 * k] * w[l + 32 * k];
#pragma unroll
        for (int off = 16; off; off >>= 1) s += __shfl_xor(s, off, 32);
        if (l == 0) sctx[gw] = s;
    } else if (gw < NC + NQ) {
        const int id = gw - NC;
        const float* row = qry + (size_t)id * DD;
        const float* w2  = w + DD;
        float s = 0.f;
#pragma unroll
        for (int k = 0; k < 8; ++k) s += row[l + 32 * k] * w2[l + 32 * k];
#pragma unroll
        for (int off = 16; off; off >>= 1) s += __shfl_xor(s, off, 32);
        if (l == 0) sqry[id] = s;
    }
}

// ---------------------------------------------------------------------------
// k1: per (b, 64-row tile of T):
//   S-strip (16x128 per wave) via v_wmma_f32_16x16x32_bf16, softmax over J,
//   write row-max(+s_ctx) for the T-softmax, second WMMA pass for c2q,
//   stream c2q / ctx*c2q into output segments 1 and 2.
// ---------------------------------------------------------------------------
__global__ __launch_bounds__(128) void k1_attn_wmma(
    const float* __restrict__ ctx, const float* __restrict__ qry,
    const float* __restrict__ w,
    const float* __restrict__ sctx, const float* __restrict__ sqry,
    float* __restrict__ mrow, float* __restrict__ out)
{
    __shared__ __bf16 Qt[128 * TPITCH];        // query^T half-slab: Qt[d][j], 33280 B
    __shared__ __bf16 aL[4 * 16 * TPITCH];     // per-wave softmax probs (bf16), 16640 B

    const int b   = blockIdx.x >> 4;           // 16 t-tiles per batch
    const int t0  = (blockIdx.x & 15) * 64;
    const int tid = threadIdx.x;
    const int wv  = tid >> 5;
    const int l   = tid & 31;
    const int ln  = l & 15;                    // N / M lane index
    const int h   = l >> 4;                    // lane half

    const float* ctxB = ctx + (size_t)b * TT * DD;
    const float* qryB = qry + (size_t)b * JJ * DD;
    const float* w3   = w + 2 * DD;
    const int tw = t0 + wv * 16;               // this wave's 16 rows

    // ---- A fragments: A[m,k] = ctx[tw+m, k] * w3[k], bf16, full K=256 ----
    v16bf afr[8];
    {
        const float* rowp = ctxB + (size_t)(tw + ln) * DD;
#pragma unroll
        for (int kt = 0; kt < 8; ++kt) {
            const int k0 = kt * 32;
#pragma unroll
            for (int i = 0; i < 8; ++i) {
                const int kl = 2 * i + 16 * h;
                f32x2 c2 = *(const f32x2*)(rowp + k0 + kl);
                f32x2 w2 = *(const f32x2*)(w3 + k0 + kl);
                afr[kt][2 * i]     = (__bf16)(c2[0] * w2[0]);
                afr[kt][2 * i + 1] = (__bf16)(c2[1] * w2[1]);
            }
        }
    }

    // ---- S-strip: 8 j-tiles x 8 k-steps of v_wmma_f32_16x16x32_bf16 ----
    v8f sacc[8];
#pragma unroll
    for (int jt = 0; jt < 8; ++jt) {
        v8f acc = {0.f, 0.f, 0.f, 0.f, 0.f, 0.f, 0.f, 0.f};
        const float* qp = qryB + (size_t)(jt * 16 + ln) * DD;
#pragma unroll
        for (int kt = 0; kt < 8; ++kt) {
            const int k0 = kt * 32 + 16 * h;
            v16bf bfr;
#pragma unroll
            for (int i = 0; i < 8; ++i) {
                f32x2 q2 = *(const f32x2*)(qp + k0 + 2 * i);
                bfr[2 * i]     = (__bf16)q2[0];
                bfr[2 * i + 1] = (__bf16)q2[1];
            }
            acc = __builtin_amdgcn_wmma_f32_16x16x32_bf16(
                false, afr[kt], false, bfr, (short)0, acc, false, false);
        }
        sacc[jt] = acc;
    }

    // ---- softmax over J (s_ctx cancels; s_qry added) ----
#pragma unroll
    for (int jt = 0; jt < 8; ++jt) {
        const float sq = sqry[b * JJ + jt * 16 + ln];
#pragma unroll
        for (int r = 0; r < 8; ++r) sacc[jt][r] += sq;
    }
    float rmax[8], rinv[8];
#pragma unroll
    for (int r = 0; r < 8; ++r) {
        float mx = sacc[0][r];
#pragma unroll
        for (int jt = 1; jt < 8; ++jt) mx = fmaxf(mx, sacc[jt][r]);
        mx = fmaxf(mx, __shfl_xor(mx, 1, 32));
        mx = fmaxf(mx, __shfl_xor(mx, 2, 32));
        mx = fmaxf(mx, __shfl_xor(mx, 4, 32));
        mx = fmaxf(mx, __shfl_xor(mx, 8, 32));
        rmax[r] = mx;
    }
#pragma unroll
    for (int r = 0; r < 8; ++r) {
        float s = 0.f;
#pragma unroll
        for (int jt = 0; jt < 8; ++jt) {
            const float p = __expf(sacc[jt][r] - rmax[r]);
            sacc[jt][r] = p;
            s += p;
        }
        s += __shfl_xor(s, 1, 32);
        s += __shfl_xor(s, 2, 32);
        s += __shfl_xor(s, 4, 32);
        s += __shfl_xor(s, 8, 32);
        rinv[r] = 1.f / s;
    }
    // row max (incl. s_ctx) for the batch softmax over T
    if (ln == 0) {
#pragma unroll
        for (int r = 0; r < 8; ++r) {
            const int t = tw + r + 8 * h;
            mrow[b * TT + t] = rmax[r] + sctx[b * TT + t];
        }
    }

    // ---- normalized probs -> LDS (bf16), per-wave region ----
    __bf16* aw = aL + wv * 16 * TPITCH;
#pragma unroll
    for (int jt = 0; jt < 8; ++jt)
#pragma unroll
        for (int r = 0; r < 8; ++r)
            aw[(r + 8 * h) * TPITCH + jt * 16 + ln] = (__bf16)(sacc[jt][r] * rinv[r]);

    // ---- A2 fragments: A2[m, k=j] = a[m, j], K = 128 -> 4 k-steps ----
    v16bf a2[4];
#pragma unroll
    for (int kt = 0; kt < 4; ++kt) {
        const int k0 = kt * 32;
#pragma unroll
        for (int i = 0; i < 8; ++i) {
            const int kl = 2 * i + 16 * h;
            bf16x2 p = *(const bf16x2*)&aw[ln * TPITCH + k0 + kl];
            a2[kt][2 * i]     = p[0];
            a2[kt][2 * i + 1] = p[1];
        }
    }

    // ---- c2q = a @ query, D processed in two halves via LDS query^T ----
#pragma unroll 1
    for (int dh = 0; dh < 2; ++dh) {
        __syncthreads();
        // build Qt[d][j] (bf16) for d in [dh*128, dh*128+128)
        for (int idx = tid; idx < 128 * 128; idx += 128) {
            const int j = idx >> 7;
            const int d = idx & 127;
            Qt[d * TPITCH + j] = (__bf16)qryB[(size_t)j * DD + dh * 128 + d];
        }
        __syncthreads();
#pragma unroll
        for (int dt = 0; dt < 8; ++dt) {
            v8f acc = {0.f, 0.f, 0.f, 0.f, 0.f, 0.f, 0.f, 0.f};
#pragma unroll
            for (int kt = 0; kt < 4; ++kt) {
                const int k0 = kt * 32 + 16 * h;
                const __bf16* qtp = Qt + (dt * 16 + ln) * TPITCH + k0;
                v16bf bfr;
#pragma unroll
                for (int i = 0; i < 8; ++i) {
                    bf16x2 p = *(const bf16x2*)(qtp + 2 * i);
                    bfr[2 * i]     = p[0];
                    bfr[2 * i + 1] = p[1];
                }
                acc = __builtin_amdgcn_wmma_f32_16x16x32_bf16(
                    false, a2[kt], false, bfr, (short)0, acc, false, false);
            }
            // stream into output: seg1 = c2q, seg2 = ctx*c2q
            const int d = dh * 128 + dt * 16 + ln;
#pragma unroll
            for (int r = 0; r < 8; ++r) {
                const int t = tw + r + 8 * h;
                const size_t row = (size_t)(b * TT + t) * (4 * DD);
                const float y = acc[r];
                const float x = ctxB[(size_t)t * DD + d];
                out[row + DD + d]     = y;
                out[row + 2 * DD + d] = x * y;
            }
        }
    }
}

// ---------------------------------------------------------------------------
// k2: per batch: bweights = softmax_T(mrow) ; q2c[b,d] = sum_t bweights[t]*ctx[b,t,d]
// ---------------------------------------------------------------------------
__global__ __launch_bounds__(256) void k2_q2c(
    const float* __restrict__ ctx, const float* __restrict__ mrow,
    float* __restrict__ q2c)
{
    __shared__ float wl[TT];
    __shared__ float red[8];
    const int b   = blockIdx.x;
    const int tid = threadIdx.x;
    const int l   = tid & 31;
    const int wv  = tid >> 5;
    const float* m = mrow + b * TT;

    float v0 = m[tid], v1 = m[tid + 256], v2 = m[tid + 512], v3 = m[tid + 768];
    float mx = fmaxf(fmaxf(v0, v1), fmaxf(v2, v3));
#pragma unroll
    for (int off = 16; off; off >>= 1) mx = fmaxf(mx, __shfl_xor(mx, off, 32));
    if (l == 0) red[wv] = mx;
    __syncthreads();
    mx = red[0];
#pragma unroll
    for (int i = 1; i < 8; ++i) mx = fmaxf(mx, red[i]);

    float e0 = __expf(v0 - mx), e1 = __expf(v1 - mx);
    float e2 = __expf(v2 - mx), e3 = __expf(v3 - mx);
    float s = e0 + e1 + e2 + e3;
#pragma unroll
    for (int off = 16; off; off >>= 1) s += __shfl_xor(s, off, 32);
    __syncthreads();
    if (l == 0) red[wv] = s;
    __syncthreads();
    s = 0.f;
#pragma unroll
    for (int i = 0; i < 8; ++i) s += red[i];
    const float inv = 1.f / s;

    wl[tid] = e0 * inv; wl[tid + 256] = e1 * inv;
    wl[tid + 512] = e2 * inv; wl[tid + 768] = e3 * inv;
    __syncthreads();

    const float* cb = ctx + (size_t)b * TT * DD;
    float acc = 0.f;
#pragma unroll 4
    for (int t = 0; t < TT; ++t) acc += wl[t] * cb[(size_t)t * DD + tid];
    q2c[b * DD + tid] = acc;
}

// ---------------------------------------------------------------------------
// k3: seg0 = ctx, seg3 = ctx * q2c   (float4 per thread)
// ---------------------------------------------------------------------------
__global__ __launch_bounds__(256) void k3_assemble(
    const float* __restrict__ ctx, const float* __restrict__ q2c,
    float* __restrict__ out)
{
    const size_t gid = (size_t)blockIdx.x * 256 + threadIdx.x;
    const int    d4  = (int)(gid & 63);        // 64 float4 per D-row
    const size_t row = gid >> 6;               // b*T + t
    const int    b   = (int)(row >> 10);
    f32x4 x = *(const f32x4*)(ctx + row * DD + d4 * 4);
    f32x4 z = *(const f32x4*)(q2c + (size_t)b * DD + d4 * 4);
    f32x4 xz = x * z;
    float* o = out + row * (4 * DD) + d4 * 4;
    *(f32x4*)(o)            = x;
    *(f32x4*)(o + 3 * DD)   = xz;
}

// ---------------------------------------------------------------------------
extern "C" void kernel_launch(void* const* d_in, const int* in_sizes, int n_in,
                              void* d_out, int out_size, void* d_ws, size_t ws_size,
                              hipStream_t stream)
{
    const float* ctx = (const float*)d_in[0];   // (32,1024,256) f32
    const float* qry = (const float*)d_in[1];   // (32,128,256)  f32
    const float* w   = (const float*)d_in[2];   // (768,)        f32
    float* out = (float*)d_out;                 // (32,1024,1024) f32

    float* sctx = (float*)d_ws;                 // B*T   = 32768 floats
    float* sqry = sctx + BB * TT;               // B*J   = 4096  floats
    float* mrow = sqry + BB * JJ;               // B*T   = 32768 floats
    float* q2c  = mrow + BB * TT;               // B*D   = 8192  floats

    const int nrows = BB * TT + BB * JJ;        // 36864 rows, 8 waves/block
    k0_rowdots<<<(nrows + 7) / 8, 256, 0, stream>>>(ctx, qry, w, sctx, sqry);
    k1_attn_wmma<<<BB * (TT / 64), 128, 0, stream>>>(ctx, qry, w, sctx, sqry, mrow, out);
    k2_q2c<<<BB, 256, 0, stream>>>(ctx, mrow, q2c);
    k3_assemble<<<(BB * TT * DD / 4) / 256, 256, 0, stream>>>(ctx, q2c, out);
}
